// CLAM_SB_61598420959506
// MI455X (gfx1250) — compile-verified, matrix operand order checked by reading, not run on hardware
//
#include <hip/hip_runtime.h>
#include <hip/hip_bf16.h>
#include <math.h>

// ---------------- types ----------------
typedef __bf16 bf16_t;
typedef bf16_t bf16x8  __attribute__((ext_vector_type(8)));
typedef bf16_t bf16x16 __attribute__((ext_vector_type(16)));
typedef float  v8f     __attribute__((ext_vector_type(8)));

#define N_ROWS 100000
#define LDIM   1024
#define D1     512
#define D2     256
#define TILE_ROWS 64
#define NBLK ((N_ROWS + TILE_ROWS - 1) / TILE_ROWS)   // 1563 (last block: 32 valid rows)

// output layout: logits[2], Y_prob[2], Y_hat[1], A_raw[100000], inst_logits[32]
#define OUT_A_OFF    5
#define OUT_INST_OFF (5 + N_ROWS)

__device__ __forceinline__ v8f wmma_bf16(bf16x16 a, bf16x16 b, v8f c) {
  return __builtin_amdgcn_wmma_f32_16x16x32_bf16(false, a, false, b, (short)0, c, false, false);
}

// Load one 16-element bf16 fragment: elements [p..p+7] and [p+16..p+23]
__device__ __forceinline__ bf16x16 load_frag(const bf16_t* p) {
  union { bf16x16 v; bf16x8 h[2]; } u;
  u.h[0] = *(const bf16x8*)(p);
  u.h[1] = *(const bf16x8*)(p + 16);
  return u.v;
}

// Build a bf16 fragment from an fp32 LDS row (k-runs of 8 at koff and koff+16)
__device__ __forceinline__ bf16x16 cvt_frag(const float* rowp, int koff) {
  const float4 a0 = *(const float4*)(rowp + koff);
  const float4 a1 = *(const float4*)(rowp + koff + 4);
  const float4 b0 = *(const float4*)(rowp + koff + 16);
  const float4 b1 = *(const float4*)(rowp + koff + 20);
  bf16x16 r;
  r[0]  = (bf16_t)a0.x; r[1]  = (bf16_t)a0.y; r[2]  = (bf16_t)a0.z; r[3]  = (bf16_t)a0.w;
  r[4]  = (bf16_t)a1.x; r[5]  = (bf16_t)a1.y; r[6]  = (bf16_t)a1.z; r[7]  = (bf16_t)a1.w;
  r[8]  = (bf16_t)b0.x; r[9]  = (bf16_t)b0.y; r[10] = (bf16_t)b0.z; r[11] = (bf16_t)b0.w;
  r[12] = (bf16_t)b1.x; r[13] = (bf16_t)b1.y; r[14] = (bf16_t)b1.z; r[15] = (bf16_t)b1.w;
  return r;
}

// CDNA5 async global->LDS copy (ASYNCcnt-tracked); ldsOff = LDS byte offset
__device__ __forceinline__ void async_b128_to_lds(uint32_t ldsOff, const float* g) {
  asm volatile("global_load_async_to_lds_b128 %0, %1, off"
               :: "v"(ldsOff), "v"(g) : "memory");
}
__device__ __forceinline__ void wait_asynccnt0() {
  asm volatile("s_wait_asynccnt 0" ::: "memory");
}
__device__ __forceinline__ uint32_t lds_offset(const void* p) {
  // generic LDS address = {SHARED_BASE, offset[31:0]} -> low 32 bits are the DS offset
  return (uint32_t)(uintptr_t)p;
}

// ---------------- prep: transpose fp32 [K x C] -> bf16 [C x K] ----------------
__global__ void k_transpose_bf16(const float* __restrict__ src, bf16_t* __restrict__ dst,
                                 int K, int C) {
  long t = (long)blockIdx.x * 256 + threadIdx.x;
  if (t >= (long)K * C) return;
  int k = (int)(t / C);
  int c = (int)(t % C);
  dst[(size_t)c * K + k] = (bf16_t)src[t];
}

// ---------------- main fused kernel ----------------
// per block: 64 rows. async-staged h -> GEMM1 (WMMA bf16) -> h1 tile (LDS bf16)
// -> GEMM2 (WMMA bf16) -> gated attention A -> exp-weighted pooled partials.
__global__ __launch_bounds__(256) void k_main(
    const float* __restrict__ h,
    const float* __restrict__ b1, const float* __restrict__ ba,
    const float* __restrict__ bb, const float* __restrict__ Wc,
    const float* __restrict__ bc,
    const bf16_t* __restrict__ W1T,   // [D1][LDIM]
    const bf16_t* __restrict__ WaT,   // [D2][D1]
    const bf16_t* __restrict__ WbT,   // [D2][D1]
    float* __restrict__ Mpart,        // [NBLK][D1]
    float* __restrict__ eSumPart,     // [NBLK]
    float* __restrict__ Avals,        // [N]
    float* __restrict__ Aout)         // d_out + OUT_A_OFF
{
  __shared__ float  hsF[2][TILE_ROWS][32];   // 16 KB double-buffered fp32 k-slices
  __shared__ bf16_t h1s[TILE_ROWS][D1];      // 64 KB h1 tile (bf16)
  __shared__ float  wavePart[8][TILE_ROWS];  // per-wave attention partials
  __shared__ float  eArow[TILE_ROWS];

  const int tid   = threadIdx.x;
  const int wave  = tid >> 5;
  const int lane  = tid & 31;
  const int lrow  = lane & 15;
  const int lhalf = lane >> 4;
  const int koff  = lhalf * 8;
  const size_t tileRow = (size_t)blockIdx.x * TILE_ROWS;

  // ---------- async staging setup: each thread copies two 16B chunks per slice ----------
  const int srow   = tid >> 3;          // 0..31  (second row = srow+32)
  const int schunk = (tid & 7) * 4;     // float index within 32-float slice row
  size_t gr0 = tileRow + srow;          if (gr0 >= N_ROWS) gr0 = N_ROWS - 1;
  size_t gr1 = tileRow + srow + 32;     if (gr1 >= N_ROWS) gr1 = N_ROWS - 1;
  const float* g0 = h + gr0 * LDIM + schunk;
  const float* g1 = h + gr1 * LDIM + schunk;
  const uint32_t l0b0 = lds_offset(&hsF[0][srow][schunk]);
  const uint32_t l0b1 = lds_offset(&hsF[1][srow][schunk]);
  const uint32_t l1b0 = lds_offset(&hsF[0][srow + 32][schunk]);
  const uint32_t l1b1 = lds_offset(&hsF[1][srow + 32][schunk]);

  // ---------- GEMM1: h1 = relu(h @ W1 + b1) ----------
  v8f acc[4][4] = {};
  const int colb1 = wave * 64;

  // prologue: slice 0 -> buffer 0
  async_b128_to_lds(l0b0, g0);
  async_b128_to_lds(l1b0, g1);

  for (int s = 0; s < LDIM / 32; ++s) {
    wait_asynccnt0();        // my slice-s copies done
    __syncthreads();         // everyone's slice-s copies done AND compute s-1 finished
    if (s + 1 < LDIM / 32) { // refill the other buffer while slice s is consumed;
                             // safe: all reads of that buffer finished before this barrier
      const uint32_t d0 = ((s + 1) & 1) ? l0b1 : l0b0;
      const uint32_t d1 = ((s + 1) & 1) ? l1b1 : l1b0;
      async_b128_to_lds(d0, g0 + (size_t)(s + 1) * 32);
      async_b128_to_lds(d1, g1 + (size_t)(s + 1) * 32);
    }

    bf16x16 bf[4];
#pragma unroll
    for (int ct = 0; ct < 4; ++ct)
      bf[ct] = load_frag(W1T + (size_t)(colb1 + ct * 16 + lrow) * LDIM + s * 32 + koff);
#pragma unroll
    for (int rt = 0; rt < 4; ++rt) {
      bf16x16 af = cvt_frag(&hsF[s & 1][rt * 16 + lrow][0], koff);
#pragma unroll
      for (int ct = 0; ct < 4; ++ct)
        acc[rt][ct] = wmma_bf16(af, bf[ct], acc[rt][ct]);
    }
  }

  // epilogue1: bias + relu, store bf16 h1 tile
  __syncthreads();
#pragma unroll
  for (int rt = 0; rt < 4; ++rt)
#pragma unroll
    for (int ct = 0; ct < 4; ++ct) {
      const int col = colb1 + ct * 16 + lrow;
      const float bias = b1[col];
#pragma unroll
      for (int v = 0; v < 8; ++v) {
        float x = acc[rt][ct][v] + bias;
        x = x > 0.f ? x : 0.f;
        h1s[rt * 16 + v + 8 * lhalf][col] = (bf16_t)x;
      }
    }
  __syncthreads();

  // ---------- GEMM2: a_pre / g_pre = h1 @ Wa / Wb ----------
  v8f f2[4][2][2] = {};          // [rowtile][coltile][a|g]
  const int colb2 = wave * 32;
  for (int s = 0; s < D1 / 32; ++s) {
    bf16x16 bfa[2], bfb[2];
#pragma unroll
    for (int ct = 0; ct < 2; ++ct) {
      const size_t boff = (size_t)(colb2 + ct * 16 + lrow) * D1 + s * 32 + koff;
      bfa[ct] = load_frag(WaT + boff);
      bfb[ct] = load_frag(WbT + boff);
    }
#pragma unroll
    for (int rt = 0; rt < 4; ++rt) {
      bf16x16 af = load_frag(&h1s[rt * 16 + lrow][s * 32 + koff]);
#pragma unroll
      for (int ct = 0; ct < 2; ++ct) {
        f2[rt][ct][0] = wmma_bf16(af, bfa[ct], f2[rt][ct][0]);
        f2[rt][ct][1] = wmma_bf16(af, bfb[ct], f2[rt][ct][1]);
      }
    }
  }

  // epilogue2: att = tanh(a)*sigmoid(g); A_row = att . Wc  (deterministic reduce)
#pragma unroll
  for (int rt = 0; rt < 4; ++rt) {
    float rv[8] = {0.f, 0.f, 0.f, 0.f, 0.f, 0.f, 0.f, 0.f};
#pragma unroll
    for (int ct = 0; ct < 2; ++ct) {
      const int col = colb2 + ct * 16 + lrow;
      const float vba = ba[col], vbb = bb[col], vwc = Wc[col];
#pragma unroll
      for (int v = 0; v < 8; ++v) {
        float a = tanhf(f2[rt][ct][0][v] + vba);
        float g = 1.f / (1.f + expf(-(f2[rt][ct][1][v] + vbb)));
        rv[v] += a * g * vwc;
      }
    }
#pragma unroll
    for (int v = 0; v < 8; ++v) {
      float x = rv[v];
      x += __shfl_xor(x, 1);
      x += __shfl_xor(x, 2);
      x += __shfl_xor(x, 4);
      x += __shfl_xor(x, 8);
      if (lrow == 0) wavePart[wave][rt * 16 + v + 8 * lhalf] = x;
    }
  }
  __syncthreads();

  if (tid < TILE_ROWS) {
    float a = bc[0];
#pragma unroll
    for (int w = 0; w < 8; ++w) a += wavePart[w][tid];
    const bool valid = (tileRow + tid) < N_ROWS;
    if (valid) {
      Aout[tileRow + tid]  = a;
      Avals[tileRow + tid] = a;
    }
    eArow[tid] = valid ? expf(a) : 0.f;   // softmax shift-invariant; A is O(1) here
  }
  __syncthreads();

  // pooled partials: P[col] = sum_r exp(A_r) * h1[r][col]
  {
    const int c0 = tid * 2, c1 = c0 + 1;
    float p0 = 0.f, p1 = 0.f;
    for (int r = 0; r < TILE_ROWS; ++r) {
      const float e = eArow[r];
      p0 += e * (float)h1s[r][c0];
      p1 += e * (float)h1s[r][c1];
    }
    Mpart[(size_t)blockIdx.x * D1 + c0] = p0;
    Mpart[(size_t)blockIdx.x * D1 + c1] = p1;
  }
  if (tid == 0) {
    float se = 0.f;
    for (int r = 0; r < TILE_ROWS; ++r) se += eArow[r];
    eSumPart[blockIdx.x] = se;
  }
}

// ---------------- reduce pooled partials -> M ----------------
__global__ __launch_bounds__(256) void k_reduce(
    const float* __restrict__ Mpart, const float* __restrict__ eSumPart,
    float* __restrict__ M) {
  __shared__ float sred[256];
  const int tid = threadIdx.x;
  const int c = blockIdx.x * 256 + tid;   // 0..511
  float s = 0.f;
  for (int b = tid; b < NBLK; b += 256) s += eSumPart[b];
  sred[tid] = s;
  __syncthreads();
  for (int off = 128; off > 0; off >>= 1) {
    if (tid < off) sred[tid] += sred[tid + off];
    __syncthreads();
  }
  const float se = sred[0];
  float acc = 0.f;
  for (int b = 0; b < NBLK; ++b) acc += Mpart[(size_t)b * D1 + c];
  M[c] = acc / se;
}

// ---------------- top-k (8 max, 8 min) with lax.top_k tie-breaking ----------------
__global__ __launch_bounds__(1024) void k_topk(const float* __restrict__ A,
                                               int* __restrict__ topIdx) {
  __shared__ float bv[1024];
  __shared__ int   bi[1024];
  __shared__ int   chosen[16];
  const int tid = threadIdx.x;
  for (int pass = 0; pass < 16; ++pass) {
    const bool mx = pass < 8;
    const int lo = mx ? 0 : 8;
    float best  = mx ? -1e30f : 1e30f;
    int   besti = N_ROWS;
    for (int i = tid; i < N_ROWS; i += 1024) {
      bool skip = false;
      for (int j = lo; j < pass; ++j)
        if (chosen[j] == i) skip = true;
      if (skip) continue;
      const float v = A[i];
      const bool take = mx ? (v > best || (v == best && i < besti))
                           : (v < best || (v == best && i < besti));
      if (take) { best = v; besti = i; }
    }
    bv[tid] = best; bi[tid] = besti;
    __syncthreads();
    for (int off = 512; off > 0; off >>= 1) {
      if (tid < off) {
        const float v2 = bv[tid + off];
        const int   i2 = bi[tid + off];
        const bool take = mx ? (v2 > bv[tid] || (v2 == bv[tid] && i2 < bi[tid]))
                             : (v2 < bv[tid] || (v2 == bv[tid] && i2 < bi[tid]));
        if (take) { bv[tid] = v2; bi[tid] = i2; }
      }
      __syncthreads();
    }
    if (tid == 0) { chosen[pass] = bi[0]; topIdx[pass] = bi[0]; }
    __syncthreads();
  }
}

// ---------------- final heads: recompute 16 h1 rows (fp32), inst + bag outputs ----------------
__global__ __launch_bounds__(256) void k_final(
    const float* __restrict__ h,  const float* __restrict__ W1,
    const float* __restrict__ b1, const float* __restrict__ Wbag,
    const float* __restrict__ bbag, const float* __restrict__ Winst,
    const float* __restrict__ binst, const float* __restrict__ M,
    const int* __restrict__ topIdx, float* __restrict__ out) {
  __shared__ float h1sel[16][D1];   // 32 KB
  __shared__ float lg[2];
  const int tid = threadIdx.x;
  for (int j = tid; j < 16 * D1; j += 256) {
    const int inst = j >> 9;
    const int col  = j & (D1 - 1);
    const size_t id = (size_t)topIdx[inst];
    float acc = b1[col];
    const float* hr = h + id * LDIM;
    for (int k = 0; k < LDIM; ++k) acc += hr[k] * W1[(size_t)k * D1 + col];
    h1sel[inst][col] = acc > 0.f ? acc : 0.f;
  }
  if (tid < 2) {
    float acc = bbag[tid];
    for (int d = 0; d < D1; ++d) acc += M[d] * Wbag[d * 2 + tid];
    lg[tid] = acc;
  }
  __syncthreads();
  if (tid < 32) {
    const int inst = tid >> 1, c = tid & 1;
    float acc = binst[c];
    for (int d = 0; d < D1; ++d) acc += h1sel[inst][d] * Winst[d * 2 + c];
    out[OUT_INST_OFF + inst * 2 + c] = acc;
  }
  if (tid == 0) {
    const float l0 = lg[0], l1 = lg[1];
    out[0] = l0; out[1] = l1;
    const float m = fmaxf(l0, l1);
    const float e0 = expf(l0 - m), e1 = expf(l1 - m), s = e0 + e1;
    out[2] = e0 / s; out[3] = e1 / s;
    out[4] = (l1 > l0) ? 1.0f : 0.0f;   // argmax, first-max wins
  }
}

// ---------------- launcher ----------------
static inline size_t align256(size_t x) { return (x + 255) & ~(size_t)255; }

extern "C" void kernel_launch(void* const* d_in, const int* in_sizes, int n_in,
                              void* d_out, int out_size, void* d_ws, size_t ws_size,
                              hipStream_t stream) {
  const float* h     = (const float*)d_in[0];
  const float* W1    = (const float*)d_in[1];
  const float* b1    = (const float*)d_in[2];
  const float* Wa    = (const float*)d_in[3];
  const float* ba    = (const float*)d_in[4];
  const float* Wb    = (const float*)d_in[5];
  const float* bb    = (const float*)d_in[6];
  const float* Wc    = (const float*)d_in[7];
  const float* bc    = (const float*)d_in[8];
  const float* Wbag  = (const float*)d_in[9];
  const float* bbag  = (const float*)d_in[10];
  const float* Winst = (const float*)d_in[11];
  const float* binst = (const float*)d_in[12];
  float* out = (float*)d_out;

  // workspace carve-up
  char* ws = (char*)d_ws;
  size_t o = 0;
  bf16_t* W1T = (bf16_t*)(ws + o); o = align256(o + (size_t)D1 * LDIM * sizeof(bf16_t));
  bf16_t* WaT = (bf16_t*)(ws + o); o = align256(o + (size_t)D2 * D1 * sizeof(bf16_t));
  bf16_t* WbT = (bf16_t*)(ws + o); o = align256(o + (size_t)D2 * D1 * sizeof(bf16_t));
  float* Mpart    = (float*)(ws + o); o = align256(o + (size_t)NBLK * D1 * sizeof(float));
  float* eSumPart = (float*)(ws + o); o = align256(o + (size_t)NBLK * sizeof(float));
  float* Avals    = (float*)(ws + o); o = align256(o + (size_t)N_ROWS * sizeof(float));
  float* Mvec     = (float*)(ws + o); o = align256(o + (size_t)D1 * sizeof(float));
  int*   topIdx   = (int*)(ws + o);   o = align256(o + 16 * sizeof(int));

  // prep: bf16 transposed weights (stay L2-resident)
  k_transpose_bf16<<<((size_t)LDIM * D1 + 255) / 256, 256, 0, stream>>>(W1, W1T, LDIM, D1);
  k_transpose_bf16<<<((size_t)D1 * D2 + 255) / 256, 256, 0, stream>>>(Wa, WaT, D1, D2);
  k_transpose_bf16<<<((size_t)D1 * D2 + 255) / 256, 256, 0, stream>>>(Wb, WbT, D1, D2);

  // fused backbone + attention + pooled partials
  k_main<<<NBLK, 256, 0, stream>>>(h, b1, ba, bb, Wc, bc, W1T, WaT, WbT,
                                   Mpart, eSumPart, Avals, out + OUT_A_OFF);

  // pooled vector M
  k_reduce<<<2, 256, 0, stream>>>(Mpart, eSumPart, Mvec);

  // top-8 / bottom-8 instance selection
  k_topk<<<1, 1024, 0, stream>>>(Avals, topIdx);

  // heads
  k_final<<<1, 256, 0, stream>>>(h, W1, b1, Wbag, bbag, Winst, binst,
                                 Mvec, topIdx, out);
}
